// SimpleTransformerWithPermutation_19963007992600
// MI455X (gfx1250) — matrix operand order repaired and checked by reading, not compiled, not run on hardware
//
#include <hip/hip_runtime.h>

// ---------------------------------------------------------------------------
// Model constants (match reference)
// ---------------------------------------------------------------------------
#define DM      768
#define DFFM    3072
#define BQ      16
#define SQ      512
#define HN      8
#define DHD     96
#define MROWS   (BQ * SQ)      // 8192
#define NLAYERS 4

typedef __attribute__((ext_vector_type(16))) __bf16 v16bf;
typedef __attribute__((ext_vector_type(8)))  float  v8f;
typedef __attribute__((ext_vector_type(4)))  int    v4i;

__device__ __forceinline__ int imin(int a, int b) { return a < b ? a : b; }

__device__ __forceinline__ float wave_sum32(float v) {
#pragma unroll
    for (int off = 16; off > 0; off >>= 1) v += __shfl_xor(v, off, 32);
    return v;
}
__device__ __forceinline__ float wave_max32(float v) {
#pragma unroll
    for (int off = 16; off > 0; off >>= 1) v = fmaxf(v, __shfl_xor(v, off, 32));
    return v;
}

// ---- async global->LDS staging (CDNA5 ASYNCcnt path), with safe fallback ---
#if defined(__has_builtin)
#if __has_builtin(__builtin_amdgcn_global_load_async_to_lds_b128)
#define HAVE_ASYNC_LDS 1
#endif
#if __has_builtin(__builtin_amdgcn_s_wait_asynccnt)
#define HAVE_WAIT_ASYNC 1
#endif
#endif

__device__ __forceinline__ void stage_b128(const __bf16* gp, __bf16* lp) {
#ifdef HAVE_ASYNC_LDS
    __builtin_amdgcn_global_load_async_to_lds_b128(
        (__attribute__((address_space(1))) v4i*)gp,
        (__attribute__((address_space(3))) v4i*)lp, 0, 0);
#else
    *(uint4*)lp = *(const uint4*)gp;
#endif
}
__device__ __forceinline__ void wait_stage() {
#ifdef HAVE_ASYNC_LDS
#ifdef HAVE_WAIT_ASYNC
    __builtin_amdgcn_s_wait_asynccnt(0);
#else
    asm volatile("s_wait_asynccnt 0" ::: "memory");
#endif
#endif
}

union Frag32 {
    struct { uint4 lo, hi; } u;
    v16bf v;
};
__device__ __forceinline__ v16bf frag_pair(const __bf16* plo, const __bf16* phi) {
    Frag32 f;
    f.u.lo = *(const uint4*)plo;
    f.u.hi = *(const uint4*)phi;
    return f.v;
}

// ---------------------------------------------------------------------------
// Batched bf16 WMMA GEMM:
//   out[z][m][n] = act( alpha * sum_k A[z][m][k]*B[z][k][n] + bias[n] (+res) )
// A: bf16 [M,K]. B: bf16, either W[N,K] (bIsNK=1) or [K,N] row-major (bIsNK=0).
// out: f32 or bf16 (outBf16). Batch z = zo*nInner + zi with separate strides.
// Block: 256 thr = 8 waves; block tile 256x64, wave tile 32x64 (8 WMMA/kstep).
// B tile (64x32, pad stride 40) double-buffered in LDS via async copies.
// Requires K % 32 == 0; grid.y*256 must cover M exactly (barriers inside).
// ---------------------------------------------------------------------------
__global__ __launch_bounds__(256) void gemm_wmma_bf16(
    const __bf16* __restrict__ A,  long lda, long aOut, long aIn,
    const __bf16* __restrict__ Bm, long ldb, long bOut, long bIn, int bIsNK,
    const float* __restrict__ bias,
    const float* __restrict__ res, long ldr,
    void* __restrict__ out, long ldo, long oOut, long oIn, int outBf16,
    int nInner, int M, int N, int K, float alpha, int doRelu)
{
    __shared__ __bf16 ldsB[2][64 * 40];   // 64 rows x 32 bf16, stride 40

    const int tid   = threadIdx.x;
    const int wave  = tid >> 5;
    const int lane  = tid & 31;
    const int lmod  = lane & 15;
    const int lhalf = lane >> 4;

    const int z  = blockIdx.z;
    const int zo = z / nInner;
    const int zi = z - zo * nInner;
    A  += (long)zo * aOut + (long)zi * aIn;
    Bm += (long)zo * bOut + (long)zi * bIn;
    const long obase = (long)zo * oOut + (long)zi * oIn;

    const int m0 = blockIdx.y * 256 + wave * 32;
    const int n0 = blockIdx.x * 64;

    const int mRow0 = imin(m0 + lmod, M - 1);
    const int mRow1 = imin(m0 + 16 + lmod, M - 1);
    const __bf16* arow0 = A + (long)mRow0 * lda;
    const __bf16* arow1 = A + (long)mRow1 * lda;

    // staging indices (NK path: one async b128 per thread per k-step)
    const int srow = tid >> 2;          // 0..63 (tile row = n)
    const int sseg = tid & 3;           // 16B segment within 32-elem row
    const int srowc = imin(n0 + srow, N - 1);

    const int nsteps = K >> 5;

    v8f accs[2][4] = {};

    // ---- stage first B tile --------------------------------------------
    if (bIsNK) {
        stage_b128(Bm + (long)srowc * ldb + 0 + sseg * 8,
                   &ldsB[0][srow * 40 + sseg * 8]);
    } else {
#pragma unroll
        for (int e = 0; e < 8; ++e) {
            int idx = tid + e * 256;          // 0..2047
            int k = idx >> 6, n = idx & 63;
            ldsB[0][n * 40 + k] = Bm[(long)k * ldb + imin(n0 + n, N - 1)];
        }
    }
    wait_stage();
    __syncthreads();

    for (int s = 0; s < nsteps; ++s) {
        const int kk  = s << 5;
        const int cur = s & 1;
        const int nxt = cur ^ 1;

        // ---- prefetch next B tile into the other LDS buffer ------------
        if (s + 1 < nsteps) {
            const int kn = kk + 32;
            if (bIsNK) {
                stage_b128(Bm + (long)srowc * ldb + kn + sseg * 8,
                           &ldsB[nxt][srow * 40 + sseg * 8]);
            } else {
#pragma unroll
                for (int e = 0; e < 8; ++e) {
                    int idx = tid + e * 256;
                    int k = idx >> 6, n = idx & 63;
                    ldsB[nxt][n * 40 + k] =
                        Bm[(long)(kn + k) * ldb + imin(n0 + n, N - 1)];
                }
            }
        }

        // ---- A fragments (ISA 16-bit A 16x32 layout) --------------------
        v16bf af0 = frag_pair(arow0 + kk + lhalf * 8, arow0 + kk + 16 + lhalf * 8);
        v16bf af1 = frag_pair(arow1 + kk + lhalf * 8, arow1 + kk + 16 + lhalf * 8);

        // ---- B fragments from LDS + 8 WMMAs -----------------------------
#pragma unroll
        for (int t = 0; t < 4; ++t) {
            const __bf16* bp = &ldsB[cur][(t * 16 + lmod) * 40 + lhalf * 8];
            v16bf bf = frag_pair(bp, bp + 16);
            accs[0][t] = __builtin_amdgcn_wmma_f32_16x16x32_bf16(
                false, af0, false, bf, (short)0, accs[0][t], false, false);
            accs[1][t] = __builtin_amdgcn_wmma_f32_16x16x32_bf16(
                false, af1, false, bf, (short)0, accs[1][t], false, false);
        }

        wait_stage();
        __syncthreads();
    }

    // ---- epilogue: C 16x16 f32 layout: m = m0 + w*16 + 8*lhalf + r, n = lmod
#pragma unroll
    for (int w = 0; w < 2; ++w) {
        const int mBase = m0 + w * 16 + lhalf * 8;
#pragma unroll
        for (int t = 0; t < 4; ++t) {
            int n = n0 + t * 16 + lmod;
            if (n >= N) continue;
            float bv = bias ? bias[n] : 0.0f;
#pragma unroll
            for (int r = 0; r < 8; ++r) {
                int m = mBase + r;
                if (m >= M) continue;
                float v = accs[w][t][r] * alpha + bv;
                if (res)    v += res[(long)m * ldr + n];
                if (doRelu) v = fmaxf(v, 0.0f);
                if (outBf16) ((__bf16*)out)[obase + (long)m * ldo + n] = (__bf16)v;
                else         ((float*) out)[obase + (long)m * ldo + n] = v;
            }
        }
    }
}

// ---------------------------------------------------------------------------
// f32 -> bf16 bulk convert (weights, once per launch)
// ---------------------------------------------------------------------------
__global__ __launch_bounds__(256) void cvt_bf16_kernel(
    const float* __restrict__ in, __bf16* __restrict__ out, long n)
{
    long base = (long)blockIdx.x * 2048 + threadIdx.x;
#pragma unroll
    for (int j = 0; j < 8; ++j) {
        long i = base + (long)j * 256;
        if (i < n) out[i] = (__bf16)in[i];
    }
}

// ---------------------------------------------------------------------------
// Embedding gather: writes f32 h (for residual) and bf16 h (GEMM operand)
// ---------------------------------------------------------------------------
__global__ __launch_bounds__(256) void embed_kernel(
    const int* __restrict__ x, const float* __restrict__ emb,
    float* __restrict__ h, __bf16* __restrict__ hb)
{
    const int row = blockIdx.x;
    const long tok = x[row];
    const float* src = emb + tok * (long)DM;
    float*  dst  = h  + (long)row * DM;
    __bf16* dstb = hb + (long)row * DM;
    for (int d = threadIdx.x; d < DM; d += 256) {
        float v = src[d];
        dst[d] = v;
        dstb[d] = (__bf16)v;
    }
}

// ---------------------------------------------------------------------------
// Row softmax over 512 cols (wave per row), bf16 output (attn@V operand)
// ---------------------------------------------------------------------------
__global__ __launch_bounds__(256) void softmax512_kernel(
    const float* __restrict__ in, __bf16* __restrict__ outb, int rows)
{
    const int row = blockIdx.x * 8 + (threadIdx.x >> 5);
    if (row >= rows) return;
    const int lane = threadIdx.x & 31;
    const float* p = in + (long)row * SQ;
    __bf16* o = outb + (long)row * SQ;

    float v[16];
    float mx = -3.4e38f;
#pragma unroll
    for (int j = 0; j < 16; ++j) { v[j] = p[lane + 32 * j]; mx = fmaxf(mx, v[j]); }
    mx = wave_max32(mx);
    float s = 0.0f;
#pragma unroll
    for (int j = 0; j < 16; ++j) { v[j] = __expf(v[j] - mx); s += v[j]; }
    s = wave_sum32(s);
    const float inv = 1.0f / s;
#pragma unroll
    for (int j = 0; j < 16; ++j) o[lane + 32 * j] = (__bf16)(v[j] * inv);
}

// ---------------------------------------------------------------------------
// LayerNorm over 768 cols (wave per row); writes f32 + bf16
// ---------------------------------------------------------------------------
__global__ __launch_bounds__(256) void layernorm768_kernel(
    const float* __restrict__ in, const float* __restrict__ sc,
    const float* __restrict__ bi, float* __restrict__ out,
    __bf16* __restrict__ outb, int rows)
{
    const int row = blockIdx.x * 8 + (threadIdx.x >> 5);
    if (row >= rows) return;
    const int lane = threadIdx.x & 31;
    const float* p = in + (long)row * DM;

    float v[24];
    float s = 0.0f, s2 = 0.0f;
#pragma unroll
    for (int j = 0; j < 24; ++j) {
        v[j] = p[lane + 32 * j];
        s += v[j]; s2 += v[j] * v[j];
    }
    s = wave_sum32(s); s2 = wave_sum32(s2);
    const float mean = s * (1.0f / DM);
    const float var  = s2 * (1.0f / DM) - mean * mean;
    const float rstd = rsqrtf(var + 1e-5f);
    float*  o  = out  + (long)row * DM;
    __bf16* ob = outb + (long)row * DM;
#pragma unroll
    for (int j = 0; j < 24; ++j) {
        int d = lane + 32 * j;
        float y = (v[j] - mean) * rstd * sc[d] + bi[d];
        o[d] = y;
        ob[d] = (__bf16)y;
    }
}

// ---------------------------------------------------------------------------
// Batched row permutation of both f32 and bf16 hidden state
// ---------------------------------------------------------------------------
__global__ __launch_bounds__(256) void permute_kernel(
    const float* __restrict__ in, const __bf16* __restrict__ inb,
    const int* __restrict__ perm,
    float* __restrict__ out, __bf16* __restrict__ outb)
{
    const int s = blockIdx.x;
    const int b = blockIdx.y;
    const int src = perm[b * SQ + s];
    const long si = ((long)b * SQ + src) * DM;
    const long di = ((long)b * SQ + s)   * DM;
    for (int d = threadIdx.x; d < DM; d += 256) {
        out[di + d]  = in[si + d];
        outb[di + d] = inb[si + d];
    }
}

// ---------------------------------------------------------------------------
// Mean pool + tiny classifier
// ---------------------------------------------------------------------------
__global__ __launch_bounds__(256) void pool_kernel(
    const float* __restrict__ h, float* __restrict__ pooled)
{
    const int b = blockIdx.x;
    for (int d = threadIdx.x; d < DM; d += 256) {
        float s = 0.0f;
        const float* p = h + (long)b * SQ * DM + d;
        for (int i = 0; i < SQ; ++i) s += p[(long)i * DM];
        pooled[b * DM + d] = s * (1.0f / SQ);
    }
}

__global__ __launch_bounds__(64) void classifier_kernel(
    const float* __restrict__ pooled, const float* __restrict__ cw,
    const float* __restrict__ cb, float* __restrict__ out)
{
    const int t = threadIdx.x;
    if (t >= 32) return;
    const int b = t >> 1, c = t & 1;
    float s = cb[c];
    const float* p = pooled + b * DM;
    const float* w = cw + c * DM;
    for (int d = 0; d < DM; ++d) s += p[d] * w[d];
    out[b * 2 + c] = s;
}

// ---------------------------------------------------------------------------
// Host orchestration
// ---------------------------------------------------------------------------
extern "C" void kernel_launch(void* const* d_in, const int* in_sizes, int n_in,
                              void* d_out, int out_size, void* d_ws, size_t ws_size,
                              hipStream_t stream) {
    (void)in_sizes; (void)n_in; (void)out_size; (void)ws_size;

    const int*   x     = (const int*)  d_in[0];
    const int*   perms = (const int*)  d_in[1];
    const float* emb   = (const float*)d_in[2];
    const float* ipw   = (const float*)d_in[3];
    const float* ipb   = (const float*)d_in[4];
    const float* ow    = (const float*)d_in[5];
    const float* ob    = (const float*)d_in[6];
    const float* l1s   = (const float*)d_in[7];
    const float* l1b   = (const float*)d_in[8];
    const float* l2s   = (const float*)d_in[9];
    const float* l2b   = (const float*)d_in[10];
    const float* f1w   = (const float*)d_in[11];
    const float* f1b   = (const float*)d_in[12];
    const float* f2w   = (const float*)d_in[13];
    const float* f2b   = (const float*)d_in[14];
    const float* cw    = (const float*)d_in[15];
    const float* cb    = (const float*)d_in[16];
    float* outp = (float*)d_out;

    char* ws = (char*)d_ws;
    size_t off = 0;
    auto carve = [&](size_t bytes) -> void* {
        void* p = ws + off;
        off += (bytes + 255) & ~(size_t)255;
        return p;
    };
    float*  hA     = (float*) carve((size_t)MROWS * DM * 4);
    float*  hB     = (float*) carve((size_t)MROWS * DM * 4);
    __bf16* hAb    = (__bf16*)carve((size_t)MROWS * DM * 2);
    __bf16* hBb    = (__bf16*)carve((size_t)MROWS * DM * 2);
    __bf16* qkvb   = (__bf16*)carve((size_t)MROWS * 3 * DM * 2);
    float*  scores = (float*) carve((size_t)BQ * HN * SQ * SQ * 4);
    __bf16* probsb = (__bf16*)carve((size_t)BQ * HN * SQ * SQ * 2);
    __bf16* attnob = (__bf16*)carve((size_t)MROWS * DM * 2);
    float*  tmp    = (float*) carve((size_t)MROWS * DM * 4);
    __bf16* ffb    = (__bf16*)carve((size_t)MROWS * DFFM * 2);
    float*  pooled = (float*) carve((size_t)BQ * DM * 4);
    __bf16* ipwb   = (__bf16*)carve((size_t)NLAYERS * 3 * DM * DM * 2);
    __bf16* owb    = (__bf16*)carve((size_t)NLAYERS * DM * DM * 2);
    __bf16* f1wb   = (__bf16*)carve((size_t)NLAYERS * DFFM * DM * 2);
    __bf16* f2wb   = (__bf16*)carve((size_t)NLAYERS * DM * DFFM * 2);

    auto cvt = [&](const float* in, __bf16* out, long n) {
        cvt_bf16_kernel<<<(int)((n + 2047) / 2048), 256, 0, stream>>>(in, out, n);
    };
    cvt(ipw, ipwb, (long)NLAYERS * 3 * DM * DM);
    cvt(ow,  owb,  (long)NLAYERS * DM * DM);
    cvt(f1w, f1wb, (long)NLAYERS * DFFM * DM);
    cvt(f2w, f2wb, (long)NLAYERS * DM * DFFM);

    auto gemm = [&](const __bf16* A, long lda, long aOut, long aIn,
                    const __bf16* Bm, long ldb, long bOut, long bIn, int bIsNK,
                    const float* bias, const float* res, long ldr,
                    void* out, long ldo, long oOut, long oIn, int outBf16,
                    int nInner, int M, int N, int K, float alpha, int relu,
                    int batches) {
        dim3 grid((N + 63) / 64, (M + 255) / 256, batches);
        gemm_wmma_bf16<<<grid, 256, 0, stream>>>(
            A, lda, aOut, aIn, Bm, ldb, bOut, bIn, bIsNK,
            bias, res, ldr, out, ldo, oOut, oIn, outBf16,
            nInner, M, N, K, alpha, relu);
    };

    embed_kernel<<<MROWS, 256, 0, stream>>>(x, emb, hA, hAb);

    float*  hCur  = hA;  float*  hAlt  = hB;
    __bf16* hCurB = hAb; __bf16* hAltB = hBb;
    const float scale = 1.0f / sqrtf((float)DHD);

    for (int l = 0; l < NLAYERS; ++l) {
        const __bf16* ipw_l = ipwb + (size_t)l * 3 * DM * DM;
        const float*  ipb_l = ipb  + (size_t)l * 3 * DM;
        const __bf16* ow_l  = owb  + (size_t)l * DM * DM;
        const float*  ob_l  = ob   + (size_t)l * DM;
        const __bf16* f1w_l = f1wb + (size_t)l * DFFM * DM;
        const float*  f1b_l = f1b  + (size_t)l * DFFM;
        const __bf16* f2w_l = f2wb + (size_t)l * DM * DFFM;
        const float*  f2b_l = f2b  + (size_t)l * DM;

        // fused QKV projection -> bf16
        gemm(hCurB, DM, 0, 0, ipw_l, DM, 0, 0, 1,
             ipb_l, nullptr, 0, qkvb, 3 * DM, 0, 0, /*bf16*/1,
             1, MROWS, 3 * DM, DM, 1.0f, 0, 1);

        // scores = scale * Q @ K^T per (b,h) -> f32
        gemm(qkvb,      3 * DM, (long)SQ * 3 * DM, DHD,
             qkvb + DM, 3 * DM, (long)SQ * 3 * DM, DHD, 1,
             nullptr, nullptr, 0,
             scores, SQ, (long)HN * SQ * SQ, (long)SQ * SQ, /*f32*/0,
             HN, SQ, SQ, DHD, scale, 0, BQ * HN);

        softmax512_kernel<<<(BQ * HN * SQ) / 8, 256, 0, stream>>>(
            scores, probsb, BQ * HN * SQ);

        // attn @ V per (b,h) -> bf16 (B is [K,N] row-major -> LDS transpose)
        gemm(probsb,         SQ,     (long)HN * SQ * SQ, (long)SQ * SQ,
             qkvb + 2 * DM,  3 * DM, (long)SQ * 3 * DM,  DHD, 0,
             nullptr, nullptr, 0,
             attnob, DM, (long)SQ * DM, DHD, /*bf16*/1,
             HN, SQ, DHD, SQ, 1.0f, 0, BQ * HN);

        // out projection + residual -> f32 tmp
        gemm(attnob, DM, 0, 0, ow_l, DM, 0, 0, 1,
             ob_l, hCur, DM, tmp, DM, 0, 0, /*f32*/0,
             1, MROWS, DM, DM, 1.0f, 0, 1);

        layernorm768_kernel<<<MROWS / 8, 256, 0, stream>>>(
            tmp, l1s + l * DM, l1b + l * DM, hCur, hCurB, MROWS);

        // FF1 + relu -> bf16
        gemm(hCurB, DM, 0, 0, f1w_l, DM, 0, 0, 1,
             f1b_l, nullptr, 0, ffb, DFFM, 0, 0, /*bf16*/1,
             1, MROWS, DFFM, DM, 1.0f, 1, 1);

        // FF2 + residual -> f32 tmp
        gemm(ffb, DFFM, 0, 0, f2w_l, DFFM, 0, 0, 1,
             f2b_l, hCur, DM, tmp, DM, 0, 0, /*f32*/0,
             1, MROWS, DM, DFFM, 1.0f, 0, 1);

        layernorm768_kernel<<<MROWS / 8, 256, 0, stream>>>(
            tmp, l2s + l * DM, l2b + l * DM, hCur, hCurB, MROWS);

        if (l < NLAYERS - 1) {
            const int* perm_l = perms + (size_t)l * BQ * SQ;
            permute_kernel<<<dim3(SQ, BQ), 256, 0, stream>>>(
                hCur, hCurB, perm_l, hAlt, hAltB);
            float*  t1 = hCur;  hCur  = hAlt;  hAlt  = t1;
            __bf16* t2 = hCurB; hCurB = hAltB; hAltB = t2;
        }
    }

    pool_kernel<<<BQ, 256, 0, stream>>>(hCur, pooled);
    classifier_kernel<<<1, 64, 0, stream>>>(pooled, cw, cb, outp);
}
